// Deformation_63359357550927
// MI455X (gfx1250) — matrix operand order repaired and checked by reading, not compiled, not run on hardware
//
#include <hip/hip_runtime.h>
#include <cstddef>
#include <cstdint>

// ---------------------------------------------------------------------------
// Types for CDNA5 WMMA / TDM
// ---------------------------------------------------------------------------
typedef __attribute__((ext_vector_type(16))) _Float16 v16h;
typedef __attribute__((ext_vector_type(8)))  _Float16 v8h;
typedef __attribute__((ext_vector_type(8)))  float    v8f;
typedef __attribute__((ext_vector_type(4)))  unsigned int v4u;
typedef __attribute__((ext_vector_type(8)))  int      v8i;
typedef __attribute__((ext_vector_type(4)))  int      v4i;

#define SA 264            // LDS activation row stride in halves (256 + 8 pad)
#define ROWS 128          // rows of M per block
#define OB_STRIDE 48      // obuf row stride (floats)

#if defined(__HIP_DEVICE_COMPILE__) && __has_builtin(__builtin_amdgcn_tensor_load_to_lds)
#define HAVE_TDM 1
#else
#define HAVE_TDM 0
#endif

// Matrix order matches d_in order: sp, st, p1, p2, r1, r2, s1, s2, w1, w2, m1, m2
__device__ __constant__ int TOFF[12]   = {0, 64, 128, 256, 264, 392, 400, 528, 552, 680, 688, 816};
__device__ __constant__ int MAT_NA[12] = {256, 256, 256, 9, 256, 4, 256, 48, 256, 1, 256, 1};
__device__ __constant__ int MAT_NP[12] = {256, 256, 256, 16, 256, 16, 256, 48, 256, 16, 256, 16};
#define TOTAL_TILES 824
#define WPK_HALVES (TOTAL_TILES * 512)           // 421888 halves
#define WPK_BYTES  (WPK_HALVES * 2)              // 843776 bytes
#define BPK_FLOATS (12 * 256)

// Staged bytes per matrix (tiles * 1KB)
#define WB_IN   65536u     // 64 tiles  (input layers, K=128 N=256)
#define WB_HID  131072u    // 128 tiles (hidden layers, K=256 N=256)
#define WB_O1   8192u      // 8 tiles   (1-tile heads)
#define WB_O3   24576u     // 24 tiles  (shs head)

// ---------------------------------------------------------------------------
// Small helpers
// ---------------------------------------------------------------------------
static __device__ __forceinline__ v16h cat8(v8h lo, v8h hi) {
    return __builtin_shufflevector(lo, hi, 0,1,2,3,4,5,6,7,8,9,10,11,12,13,14,15);
}

static __device__ __forceinline__ v8h cvt8(const float* __restrict__ p) {
    float4 x = *(const float4*)p;
    float4 y = *(const float4*)(p + 4);
    v8h r;
    r[0] = (_Float16)x.x; r[1] = (_Float16)x.y; r[2] = (_Float16)x.z; r[3] = (_Float16)x.w;
    r[4] = (_Float16)y.x; r[5] = (_Float16)y.y; r[6] = (_Float16)y.z; r[7] = (_Float16)y.w;
    return r;
}

// 32B fragment load (works for global or LDS pointers)
static __device__ __forceinline__ v16h loadB(const _Float16* __restrict__ p) {
    v8h lo = *(const v8h*)p;
    v8h hi = *(const v8h*)(p + 8);
    return cat8(lo, hi);
}

static __device__ __forceinline__ v16h loadA_lds(const _Float16* __restrict__ row, int k0) {
    v8h lo = *(const v8h*)(row + k0);
    v8h hi = *(const v8h*)(row + k0 + 16);
    return cat8(lo, hi);
}

static __device__ __forceinline__ v8f wmma_f16(v16h a, v16h b, v8f c) {
    return __builtin_amdgcn_wmma_f32_16x16x32_f16(false, a, false, b, (short)0, c, false, false);
}

// ---------------------------------------------------------------------------
// TDM staging: DMA a contiguous packed-weight block (global -> LDS panel).
// D#: count=1, type=2 (image), data_size=8B, 1-D tile of `ne` elements.
// Issued by wave 0 only (TDM ignores EXEC; one DMA per instruction).
// ---------------------------------------------------------------------------
static __device__ __forceinline__ void stage_weights(
    const _Float16* __restrict__ gsrc, _Float16* __restrict__ wlds,
    unsigned nbytes, int tid, int wave)
{
    __syncthreads();                       // everyone done with previous panel
#if HAVE_TDM
    if (wave == 0) {
        unsigned long long ga = (unsigned long long)(uintptr_t)gsrc;
        unsigned lds_off = (unsigned)(uintptr_t)wlds;   // LDS byte offset (low 32 bits)
        unsigned ne = nbytes >> 3;                      // 8-byte elements
        v4u g0 = { 1u,                                  // count=1 (valid user D#)
                   lds_off,                             // lds_addr
                   (unsigned)ga,                        // global_addr[31:0]
                   (unsigned)(ga >> 32) | (2u << 30) }; // global_addr[56:32] | type=2
        v8i g1 = { (int)(3u << 16),                     // workgroup_mask=0, data_size=8B
                   (int)((ne & 0xFFFFu) << 16),         // tensor_dim0[15:0]
                   (int)(ne >> 16),                     // tensor_dim0[31:16]
                   (int)((ne & 0xFFFFu) << 16),         // tile_dim0 (tensor_dim1=0)
                   0,                                   // tile_dim1=0, tile_dim2=0
                   (int)ne,                             // tensor_dim0_stride[31:0]
                   0, 0 };
        v4i gz = { 0, 0, 0, 0 };
#if __clang_major__ >= 23
        v8i gz8 = { 0, 0, 0, 0, 0, 0, 0, 0 };
        __builtin_amdgcn_tensor_load_to_lds(g0, g1, gz, gz, gz8, 0);
#else
        __builtin_amdgcn_tensor_load_to_lds(g0, g1, gz, gz, 0);
#endif
        __builtin_amdgcn_s_wait_tensorcnt(0);
    }
#else
    const v8h* s = (const v8h*)gsrc;
    v8h* d = (v8h*)wlds;
    int nv = (int)(nbytes >> 4);
    for (int i = tid; i < nv; i += 256) d[i] = s[i];
#endif
    __syncthreads();                       // panel visible to all waves
}

// ---------------------------------------------------------------------------
// Layer helpers (each wave handles a 16-row slice; A per ISA 16-bit 16x32 layout)
// Weights are read from the LDS panel (staged by TDM).
// ---------------------------------------------------------------------------

// Input layer: K=128 from global f32 grid, N=256, ReLU, dst = f16 LDS
static __device__ __forceinline__ void layer_in(
    const float* __restrict__ G, int Mrows, int growBase,
    const _Float16* __restrict__ Wt, const float* __restrict__ bias,
    _Float16* __restrict__ dst, int r0, int lm, int lg, int lane)
{
    int row = growBase + lm;
    if (row >= Mrows) row = Mrows - 1;
    const float* gp = G + (size_t)row * 128;
    v16h a[4];
#pragma unroll
    for (int kc = 0; kc < 4; ++kc) {
        int k0 = kc * 32 + lg * 8;
        a[kc] = cat8(cvt8(gp + k0), cvt8(gp + k0 + 16));
    }
    for (int nt = 0; nt < 16; ++nt) {
        v8f acc = {0.f,0.f,0.f,0.f,0.f,0.f,0.f,0.f};
#pragma unroll
        for (int kc = 0; kc < 4; ++kc) {
            v16h b = loadB(Wt + (size_t)(kc * 16 + nt) * 512 + lane * 16);
            acc = wmma_f16(a[kc], b, acc);
        }
        float bb = bias[nt * 16 + lm];
        int mrow = r0 + lg * 8;
#pragma unroll
        for (int r = 0; r < 8; ++r) {
            float v = acc[r] + bb;
            v = fmaxf(v, 0.f);
            dst[(size_t)(mrow + r) * SA + nt * 16 + lm] = (_Float16)v;
        }
    }
}

// Hidden layer: K=256 LDS f16 -> N=256 LDS f16, ReLU
static __device__ __forceinline__ void layer_hidden(
    const _Float16* __restrict__ src,
    const _Float16* __restrict__ Wt, const float* __restrict__ bias,
    _Float16* __restrict__ dst, int r0, int lm, int lg, int lane)
{
    const _Float16* arow = src + (size_t)(r0 + lm) * SA;
    v16h a[8];
#pragma unroll
    for (int kc = 0; kc < 8; ++kc) a[kc] = loadA_lds(arow, kc * 32 + lg * 8);
    for (int nt = 0; nt < 16; ++nt) {
        v8f acc = {0.f,0.f,0.f,0.f,0.f,0.f,0.f,0.f};
#pragma unroll
        for (int kc = 0; kc < 8; ++kc) {
            v16h b = loadB(Wt + (size_t)(kc * 16 + nt) * 512 + lane * 16);
            acc = wmma_f16(a[kc], b, acc);
        }
        float bb = bias[nt * 16 + lm];
        int mrow = r0 + lg * 8;
#pragma unroll
        for (int r = 0; r < 8; ++r) {
            float v = acc[r] + bb;
            v = fmaxf(v, 0.f);
            dst[(size_t)(mrow + r) * SA + nt * 16 + lm] = (_Float16)v;
        }
    }
}

// Output head layer: K=256 LDS f16 -> ntOut 16-col tiles of f32 in obuf, no ReLU
static __device__ __forceinline__ void layer_out(
    const _Float16* __restrict__ src,
    const _Float16* __restrict__ Wt, const float* __restrict__ bias,
    float* __restrict__ obuf, int ntOut, int r0, int lm, int lg, int lane)
{
    const _Float16* arow = src + (size_t)(r0 + lm) * SA;
    v16h a[8];
#pragma unroll
    for (int kc = 0; kc < 8; ++kc) a[kc] = loadA_lds(arow, kc * 32 + lg * 8);
    for (int nt = 0; nt < ntOut; ++nt) {
        v8f acc = {0.f,0.f,0.f,0.f,0.f,0.f,0.f,0.f};
#pragma unroll
        for (int kc = 0; kc < 8; ++kc) {
            v16h b = loadB(Wt + (size_t)(kc * ntOut + nt) * 512 + lane * 16);
            acc = wmma_f16(a[kc], b, acc);
        }
        float bb = bias[nt * 16 + lm];
        int mrow = r0 + lg * 8;
#pragma unroll
        for (int r = 0; r < 8; ++r)
            obuf[(mrow + r) * OB_STRIDE + nt * 16 + lm] = acc[r] + bb;
    }
}

// ---------------------------------------------------------------------------
// Kernel 0: pack weights f32 -> f16 WMMA B-fragment layout + biases (f32)
// B layout: tile (kt,nt) of 32x16; lane = ((k%16) >= 8)*16 + (n%16); half = (k%32 >= 16)*8 + k%8
// ---------------------------------------------------------------------------
struct WPtrs { const float* W[12]; const float* B[12]; };

__global__ void pack_weights(WPtrs p, _Float16* __restrict__ wpk, float* __restrict__ bpk)
{
    const int total_w = WPK_HALVES;
    const int total   = total_w + BPK_FLOATS;
    for (int i = blockIdx.x * blockDim.x + threadIdx.x; i < total;
         i += gridDim.x * blockDim.x) {
        if (i < total_w) {
            int tile   = i >> 9;
            int within = i & 511;
            int m = 0;
#pragma unroll
            for (int j = 1; j < 12; ++j) if (tile >= TOFF[j]) m = j;
            int lt  = tile - TOFF[m];
            int ntN = MAT_NP[m] >> 4;
            int kt  = lt / ntN, nt = lt % ntN;
            int ln  = within >> 4, half = within & 15;
            int k   = kt * 32 + ((half >> 3) << 4) + ((ln >> 4) << 3) + (half & 7);
            int n   = nt * 16 + (ln & 15);
            float v = 0.f;
            if (n < MAT_NA[m]) v = p.W[m][(size_t)k * MAT_NA[m] + n];
            wpk[i] = (_Float16)v;
        } else {
            int j = i - total_w;
            int m = j >> 8, n = j & 255;
            bpk[j] = (n < MAT_NA[m]) ? p.B[m][n] : 0.f;
        }
    }
}

// ---------------------------------------------------------------------------
// Kernel 1: initialize output segments from the embeddings
// out layout: pts[N*3] | rot[N*4] | opacity[N*1] | shs[N*48]
// ---------------------------------------------------------------------------
__global__ void copy_init(const float* __restrict__ pts_in, const float* __restrict__ rot_in,
                          const float* __restrict__ h_in, const float* __restrict__ shs_in,
                          float* __restrict__ out, int N)
{
    const size_t n3 = (size_t)N * 3, n7 = (size_t)N * 7, n8 = (size_t)N * 8;
    const size_t total = (size_t)N * 56;
    for (size_t i = (size_t)blockIdx.x * blockDim.x + threadIdx.x; i < total;
         i += (size_t)gridDim.x * blockDim.x) {
        float v;
        if      (i < n3) v = pts_in[i];
        else if (i < n7) v = rot_in[i - n3];
        else if (i < n8) v = h_in[i - n7];
        else             v = shs_in[i - n8];
        out[i] = v;
    }
}

// ---------------------------------------------------------------------------
// Kernel 2: fused deformation MLP, 128 rows of M per block, 8 wave32s.
// Per layer: TDM-stage the packed weight panel into LDS, then WMMA from LDS.
// ---------------------------------------------------------------------------
__global__ __launch_bounds__(256)
void fused_deform(const float* __restrict__ grid_sp, const float* __restrict__ grid_st,
                  const float* __restrict__ time_emb, const float* __restrict__ h_emb,
                  const int* __restrict__ idx,
                  const _Float16* __restrict__ Wpk, const float* __restrict__ Bpk,
                  float* __restrict__ out_pts, float* __restrict__ out_rot,
                  float* __restrict__ out_op, float* __restrict__ out_shs, int Mrows)
{
    extern __shared__ char smem_raw[];
    _Float16* actx = (_Float16*)smem_raw;                 // ROWS x SA halves
    _Float16* acth = actx + (size_t)ROWS * SA;            // ROWS x SA halves
    float*    obuf = (float*)(acth + (size_t)ROWS * SA);  // ROWS x 48 f32
    float*    wbuf = obuf + ROWS * OB_STRIDE;             // ROWS f32
    _Float16* wlds = (_Float16*)(wbuf + ROWS);            // 128KB weight panel

    const int tid = threadIdx.x;
    const int lane = tid & 31, wave = tid >> 5;
    const int lm = lane & 15, lg = lane >> 4;
    const int r0 = wave * 16;
    const int tileBase = blockIdx.x * ROWS;
    const float t = time_emb[0];

#define WMAT(m) (Wpk + (size_t)TOFF[m] * 512)
#define BVEC(m) (Bpk + (m) * 256)

    // ---- act_x = relu(grid_sp @ W_sp + b_sp)
    stage_weights(WMAT(0), wlds, WB_IN, tid, wave);
    layer_in(grid_sp, Mrows, tileBase + r0, wlds, BVEC(0), actx, r0, lm, lg, lane);
    __syncthreads();

    // ---- head P: dx -> Bezier delta -> pts scatter-add
    stage_weights(WMAT(2), wlds, WB_HID, tid, wave);
    layer_hidden(actx, wlds, BVEC(2), acth, r0, lm, lg, lane);
    stage_weights(WMAT(3), wlds, WB_O1, tid, wave);
    layer_out(acth, wlds, BVEC(3), obuf, 1, r0, lm, lg, lane);
    __syncthreads();
    if (tid < ROWS) {
        int gr = tileBase + tid;
        if (gr < Mrows) {
            int tg = idx[gr];
            float mint = 1.f - t;
            float c0 = 3.f * mint * mint * t, c1 = 3.f * mint * t * t, c2 = t * t * t;
            const float* ob = obuf + tid * OB_STRIDE;
#pragma unroll
            for (int j = 0; j < 3; ++j) {
                float d = c0 * ob[j] + c1 * ob[3 + j] + c2 * ob[6 + j];
                out_pts[(size_t)tg * 3 + j] += d;
            }
        }
    }

    // ---- head W: w = head(sp)^2 -> wbuf
    stage_weights(WMAT(8), wlds, WB_HID, tid, wave);
    layer_hidden(actx, wlds, BVEC(8), acth, r0, lm, lg, lane);
    stage_weights(WMAT(9), wlds, WB_O1, tid, wave);
    layer_out(acth, wlds, BVEC(9), obuf, 1, r0, lm, lg, lane);
    __syncthreads();
    if (tid < ROWS) {
        float wv = obuf[tid * OB_STRIDE];
        wbuf[tid] = wv * wv;
    }

    // ---- head M: mu = sigmoid(head(sp)); opacity = h * exp(-w*(t-mu)^2)
    stage_weights(WMAT(10), wlds, WB_HID, tid, wave);
    layer_hidden(actx, wlds, BVEC(10), acth, r0, lm, lg, lane);
    stage_weights(WMAT(11), wlds, WB_O1, tid, wave);
    layer_out(acth, wlds, BVEC(11), obuf, 1, r0, lm, lg, lane);
    __syncthreads();
    if (tid < ROWS) {
        int gr = tileBase + tid;
        if (gr < Mrows) {
            int tg = idx[gr];
            float mu = 1.f / (1.f + expf(-obuf[tid * OB_STRIDE]));
            float wv = wbuf[tid];
            float dd = t - mu;
            out_op[tg] = h_emb[tg] * expf(-wv * dd * dd);
        }
    }

    // ---- act_x = relu(grid_st @ W_st + b_st)  (overwrite)
    stage_weights(WMAT(1), wlds, WB_IN, tid, wave);
    layer_in(grid_st, Mrows, tileBase + r0, wlds, BVEC(1), actx, r0, lm, lg, lane);
    __syncthreads();

    // ---- head R: rotations scatter-add
    stage_weights(WMAT(4), wlds, WB_HID, tid, wave);
    layer_hidden(actx, wlds, BVEC(4), acth, r0, lm, lg, lane);
    stage_weights(WMAT(5), wlds, WB_O1, tid, wave);
    layer_out(acth, wlds, BVEC(5), obuf, 1, r0, lm, lg, lane);
    __syncthreads();
    if (tid < ROWS) {
        int gr = tileBase + tid;
        if (gr < Mrows) {
            int tg = idx[gr];
            const float* ob = obuf + tid * OB_STRIDE;
#pragma unroll
            for (int c = 0; c < 4; ++c) out_rot[(size_t)tg * 4 + c] += ob[c];
        }
    }

    // ---- head S: shs scatter-add (48 cols = 3 tiles)
    stage_weights(WMAT(6), wlds, WB_HID, tid, wave);
    layer_hidden(actx, wlds, BVEC(6), acth, r0, lm, lg, lane);
    stage_weights(WMAT(7), wlds, WB_O3, tid, wave);
    layer_out(acth, wlds, BVEC(7), obuf, 3, r0, lm, lg, lane);
    __syncthreads();
    if (tid < ROWS) {
        int gr = tileBase + tid;
        if (gr < Mrows) {
            int tg = idx[gr];
            const float* ob = obuf + tid * OB_STRIDE;
#pragma unroll 8
            for (int c = 0; c < 48; ++c) out_shs[(size_t)tg * 48 + c] += ob[c];
        }
    }
#undef WMAT
#undef BVEC
}

// ---------------------------------------------------------------------------
// Host launcher
// ---------------------------------------------------------------------------
extern "C" void kernel_launch(void* const* d_in, const int* in_sizes, int n_in,
                              void* d_out, int out_size, void* d_ws, size_t ws_size,
                              hipStream_t stream)
{
    (void)n_in; (void)out_size; (void)ws_size;
    const int N = in_sizes[0] / 3;          // rays_pts_emb is (N,3)
    const int M = in_sizes[5] / 128;        // grid_sp is (M,128)

    const float* rays     = (const float*)d_in[0];
    const float* rots     = (const float*)d_in[1];
    const float* shs      = (const float*)d_in[2];
    const float* time_emb = (const float*)d_in[3];
    const float* h_emb    = (const float*)d_in[4];
    const float* grid_sp  = (const float*)d_in[5];
    const float* grid_st  = (const float*)d_in[6];
    const int*   idx      = (const int*)d_in[7];

    WPtrs P;
    for (int m = 0; m < 12; ++m) {
        P.W[m] = (const float*)d_in[8 + 2 * m];
        P.B[m] = (const float*)d_in[9 + 2 * m];
    }

    _Float16* wpk = (_Float16*)d_ws;
    float*    bpk = (float*)((char*)d_ws + WPK_BYTES);

    float* out     = (float*)d_out;
    float* out_pts = out;
    float* out_rot = out + (size_t)N * 3;
    float* out_op  = out + (size_t)N * 7;
    float* out_shs = out + (size_t)N * 8;

    pack_weights<<<512, 256, 0, stream>>>(P, wpk, bpk);
    copy_init<<<4096, 256, 0, stream>>>(rays, rots, h_emb, shs, out, N);

    const size_t smem = (size_t)2 * ROWS * SA * sizeof(_Float16)   // activations
                      + (size_t)ROWS * OB_STRIDE * sizeof(float)   // obuf
                      + (size_t)ROWS * sizeof(float)               // wbuf
                      + (size_t)WB_HID;                            // weight panel
    hipFuncSetAttribute((const void*)fused_deform,
                        hipFuncAttributeMaxDynamicSharedMemorySize, (int)smem);

    fused_deform<<<(M + ROWS - 1) / ROWS, 256, smem, stream>>>(
        grid_sp, grid_st, time_emb, h_emb, idx, wpk, bpk,
        out_pts, out_rot, out_op, out_shs, M);
}
